// GraphTextModel_2293512536546
// MI455X (gfx1250) — compile-verified, matrix operand order checked by reading, not run on hardware
//
#include <hip/hip_runtime.h>
#include <hip/hip_bf16.h>
#include <math.h>

typedef __attribute__((ext_vector_type(16))) _Float16 v16h;
typedef __attribute__((ext_vector_type(8)))  float    v8f;

#define REG1f 0.001f
#define REG2f 0.001f
#define NSTATB 256   // blocks used by k_stats (partial-sum count)

// ---------------------------------------------------------------- helpers ---
__device__ __forceinline__ v8f wmma_f16(v16h a, v16h b, v8f c) {
  // D = A(16x32 f16) x B(32x16 f16) + C(16x16 f32)
  return __builtin_amdgcn_wmma_f32_16x16x32_f16(false, a, false, b,
                                                (short)0, c, false, false);
}

__device__ __forceinline__ void load16(float f[16], const float* __restrict__ p) {
  const float4* p4 = (const float4*)p;
#pragma unroll
  for (int i = 0; i < 4; ++i) {
    float4 v = p4[i];
    f[4*i+0] = v.x; f[4*i+1] = v.y; f[4*i+2] = v.z; f[4*i+3] = v.w;
  }
}

__device__ __forceinline__ void load16_guard(float f[16], const float* __restrict__ row,
                                             int k0, int kmax) {
  if (k0 + 16 <= kmax) {
    load16(f, row + k0);
  } else {
#pragma unroll
    for (int i = 0; i < 16; ++i) {
      int k = k0 + i;
      f[i] = (k < kmax) ? row[k] : 0.0f;
    }
  }
}

// f32 -> f16 hi/lo split: x ~= hi + lo with ~22-bit effective mantissa
__device__ __forceinline__ void split16(const float f[16], v16h& hi, v16h& lo) {
#pragma unroll
  for (int i = 0; i < 16; ++i) {
    _Float16 h = (_Float16)f[i];
    hi[i] = h;
    lo[i] = (_Float16)(f[i] - (float)h);
  }
}

// ------------------------------------------------------------- kernel 1 -----
// g = relu(X @ W1^T + b1); G = g @ W2^T + b2     (X: m x 1024, G: m x 20)
// B (=W1 slab) is pre-split to f16 hi/lo in LDS once per 128-wide K-slab and
// stored in WMMA B-fragment order: frag(c,t,lane) -> 16 halves contiguous.
__global__ __launch_bounds__(256) void k_graph(
    const float* __restrict__ X, const float* __restrict__ W1,
    const float* __restrict__ b1, const float* __restrict__ W2,
    const float* __restrict__ b2, float* __restrict__ G) {
  __shared__ __align__(32) char smem[34816];       // B slab (32KB) / g tiles (34KB)
  _Float16* bhp = (_Float16*)smem;                 // 4*4*32 frags * 16 halves
  _Float16* blp = (_Float16*)(smem + 16384);
  const int tid     = threadIdx.x;
  const int lane    = tid & 31;
  const int wave    = tid >> 5;
  const int rowBase = blockIdx.x * 128 + wave * 16;
  const int mrow    = rowBase + (lane & 15);
  const int khalf   = (lane >> 4) * 16;            // lanes 16-31 carry K+16..31
  const float* arow = X + (size_t)mrow * 1024;

  v8f acc[4] = {};
  for (int kb = 0; kb < 1024; kb += 128) {
    __syncthreads();                               // prior slab fully consumed
    // cooperative split of W1[:, kb:kb+128] into fragment-ordered f16 hi/lo
    for (int idx = tid; idx < 64 * 128; idx += 256) {
      int n = idx >> 7, j = idx & 127;
      float v = W1[(size_t)n * 1024 + kb + j];
      _Float16 h = (_Float16)v;
      int c = j >> 5, hf = (j >> 4) & 1, i = j & 15;
      int off = ((((c * 4) + (n >> 4)) * 32) + hf * 16 + (n & 15)) * 16 + i;
      bhp[off] = h;
      blp[off] = (_Float16)(v - (float)h);
    }
    __syncthreads();
    if (kb + 128 < 1024)                           // pull next A slab early
      __builtin_prefetch((const void*)(arow + kb + 128 + khalf), 0, 3);
#pragma unroll
    for (int c = 0; c < 4; ++c) {
      int kc = kb + c * 32;
      float af[16];
      load16(af, arow + kc + khalf);
      v16h ah, al; split16(af, ah, al);
#pragma unroll
      for (int t = 0; t < 4; ++t) {
        int fo = (((c * 4 + t) * 32) + lane) * 16;
        v16h bh = *(const v16h*)(bhp + fo);
        v16h bl = *(const v16h*)(blp + fo);
        acc[t] = wmma_f16(ah, bh, acc[t]);
        acc[t] = wmma_f16(ah, bl, acc[t]);
        acc[t] = wmma_f16(al, bh, acc[t]);
      }
    }
  }
  __syncthreads();                                 // done with B slab; reuse pool
  // bias + relu, stage g tile to LDS (row-major, stride 68 to dodge conflicts)
  float* gl = (float*)smem + wave * (16 * 68);
  const int m0 = (lane < 16) ? 0 : 8;
#pragma unroll
  for (int t = 0; t < 4; ++t) {
    int n = t * 16 + (lane & 15);
    float bias = b1[n];
#pragma unroll
    for (int i = 0; i < 8; ++i) {
      float v = acc[t][i] + bias;
      gl[(m0 + i) * 68 + n] = v > 0.0f ? v : 0.0f;
    }
  }
  __syncthreads();
  // tiny GEMM2 (K=64, N=20) in exact f32
  for (int e = lane; e < 16 * 20; e += 32) {
    int r = e / 20, n = e - r * 20;
    float s = b2[n];
    const float* w2r = W2 + n * 64;
#pragma unroll 8
    for (int k = 0; k < 64; ++k) s = fmaf(gl[r * 68 + k], w2r[k], s);
    G[(size_t)(rowBase + r) * 20 + n] = s;
  }
}

// ------------------------------------------------------------- kernel 2 -----
// gates = emb[ctx[:,0]] @ W_ih^T + (b_ih+b_hh);  H = sig(og)*tanh(sig(ig)*tanh(gg))
__global__ __launch_bounds__(256) void k_lstm(
    const float* __restrict__ EMB, const float* __restrict__ Wih,
    const float* __restrict__ bih, const float* __restrict__ bhh,
    const int* __restrict__ ctx, int L, float* __restrict__ H) {
  __shared__ __align__(32) char smem[43008];       // B slab (40KB) / gates (42KB)
  _Float16* bhp = (_Float16*)smem;                 // 4*5*32 frags * 16 halves
  _Float16* blp = (_Float16*)(smem + 20480);
  const int tid     = threadIdx.x;
  const int lane    = tid & 31;
  const int wave    = tid >> 5;
  const int rowBase = blockIdx.x * 128 + wave * 16;
  const int mrow    = rowBase + (lane & 15);
  const int khalf   = (lane >> 4) * 16;
  const int vrow    = ctx[(size_t)mrow * L];       // context[:, 0]
  const float* arow = EMB + (size_t)vrow * 300;

  v8f acc[5] = {};
  for (int kb = 0; kb < 384; kb += 128) {          // K=300 padded to 384
    __syncthreads();
    for (int idx = tid; idx < 80 * 128; idx += 256) {
      int n = idx >> 7, j = idx & 127;
      int k = kb + j;
      float v = (k < 300) ? Wih[(size_t)n * 300 + k] : 0.0f;
      _Float16 h = (_Float16)v;
      int c = j >> 5, hf = (j >> 4) & 1, i = j & 15;
      int off = ((((c * 5) + (n >> 4)) * 32) + hf * 16 + (n & 15)) * 16 + i;
      bhp[off] = h;
      blp[off] = (_Float16)(v - (float)h);
    }
    __syncthreads();
#pragma unroll
    for (int c = 0; c < 4; ++c) {
      int kc = kb + c * 32;
      if (kc >= 300) break;                        // uniform: skip all-pad chunks
      float af[16];
      load16_guard(af, arow, kc + khalf, 300);
      v16h ah, al; split16(af, ah, al);
#pragma unroll
      for (int t = 0; t < 5; ++t) {
        int fo = (((c * 5 + t) * 32) + lane) * 16;
        v16h bh = *(const v16h*)(bhp + fo);
        v16h bl = *(const v16h*)(blp + fo);
        acc[t] = wmma_f16(ah, bh, acc[t]);
        acc[t] = wmma_f16(ah, bl, acc[t]);
        acc[t] = wmma_f16(al, bh, acc[t]);
      }
    }
  }
  __syncthreads();                                 // reuse pool for gates tiles
  float* gl = (float*)smem + wave * (16 * 84);
  const int m0 = (lane < 16) ? 0 : 8;
#pragma unroll
  for (int t = 0; t < 5; ++t) {
    int n = t * 16 + (lane & 15);
    float bias = bih[n] + bhh[n];
#pragma unroll
    for (int i = 0; i < 8; ++i) gl[(m0 + i) * 84 + n] = acc[t][i] + bias;
  }
  __syncthreads();
  for (int e = lane; e < 16 * 20; e += 32) {
    int r = e / 20, j = e - r * 20;
    float ig = gl[r * 84 + j];
    float gg = gl[r * 84 + 40 + j];
    float og = gl[r * 84 + 60 + j];
    float si = 1.0f / (1.0f + expf(-ig));
    float so = 1.0f / (1.0f + expf(-og));
    float c1 = si * tanhf(gg);
    H[(size_t)(rowBase + r) * 20 + j] = so * tanhf(c1);
  }
}

// ------------------------------------------------------------- kernel 3 -----
// ft_out = H @ Wf^T + bf   (K=20 -> exact f32 VALU; write bound anyway)
__global__ __launch_bounds__(256) void k_ft(
    const float* __restrict__ H, const float* __restrict__ Wf,
    const float* __restrict__ bfv, float* __restrict__ out) {
  int idx = blockIdx.x * 256 + threadIdx.x;
  int row = idx >> 6;
  int col = idx & 63;
  const float* h = H + (size_t)row * 20;
  const float* w = Wf + col * 20;
  float s = bfv[col];
#pragma unroll
  for (int k = 0; k < 20; ++k) s = fmaf(h[k], w[k], s);
  out[(size_t)row * 64 + col] = s;
}

// ------------------------------------------------------------- kernel 4 -----
// per-block partials: [sum_g(20) | sum_f(20) | Ggg(400) | Gff(400) | Ggf(400)]
__global__ __launch_bounds__(256) void k_stats(
    const float* __restrict__ G, const float* __restrict__ H,
    float* __restrict__ P, int rowsPerBlock) {
  __shared__ float sg[64][20];
  __shared__ float sh[64][20];
  const int tid  = threadIdx.x;
  const int base = blockIdx.x * rowsPerBlock;
  float local[5] = {0, 0, 0, 0, 0};
  for (int r0 = 0; r0 < rowsPerBlock; r0 += 64) {
    __syncthreads();
    for (int i = tid; i < 64 * 20; i += 256) {
      int r = i / 20, c = i - r * 20;
      sg[r][c] = G[(size_t)(base + r0 + r) * 20 + c];
      sh[r][c] = H[(size_t)(base + r0 + r) * 20 + c];
    }
    __syncthreads();
#pragma unroll
    for (int u = 0; u < 5; ++u) {
      int e = tid + u * 256;
      if (e < 1240) {
        float s = local[u];
        if (e < 20)       { for (int r = 0; r < 64; ++r) s += sg[r][e]; }
        else if (e < 40)  { int j = e - 20;  for (int r = 0; r < 64; ++r) s += sh[r][j]; }
        else if (e < 440) { int i2 = (e - 40) / 20,  j = (e - 40) % 20;
                            for (int r = 0; r < 64; ++r) s = fmaf(sg[r][i2], sg[r][j], s); }
        else if (e < 840) { int i2 = (e - 440) / 20, j = (e - 440) % 20;
                            for (int r = 0; r < 64; ++r) s = fmaf(sh[r][i2], sh[r][j], s); }
        else              { int i2 = (e - 840) / 20, j = (e - 840) % 20;
                            for (int r = 0; r < 64; ++r) s = fmaf(sg[r][i2], sh[r][j], s); }
        local[u] = s;
      }
    }
  }
#pragma unroll
  for (int u = 0; u < 5; ++u) {
    int e = tid + u * 256;
    if (e < 1240) P[(size_t)blockIdx.x * 1240 + e] = local[u];
  }
}

// ------------------------------------------------------------- kernel 5 -----
// single-wave solver: covariances -> 3x Jacobi eigensolves -> nuclear norm
__device__ void jacobi20(float* A, float* V, int lane) {
  if (lane < 20) for (int i = 0; i < 20; ++i) V[i * 20 + lane] = (i == lane) ? 1.0f : 0.0f;
  __syncthreads();
  for (int sweep = 0; sweep < 12; ++sweep) {
    for (int p = 0; p < 19; ++p) {
      for (int q = p + 1; q < 20; ++q) {
        float apq = A[p * 20 + q];
        float app = A[p * 20 + p];
        float aqq = A[q * 20 + q];
        float c = 1.0f, s = 0.0f;
        if (fabsf(apq) > 1e-18f) {
          float tau = (aqq - app) / (2.0f * apq);
          float t = (tau >= 0.0f ? 1.0f : -1.0f) / (fabsf(tau) + sqrtf(1.0f + tau * tau));
          c = 1.0f / sqrtf(1.0f + t * t);
          s = t * c;
        }
        int k = lane;
        float akp = 0, akq = 0;
        if (k < 20) { akp = A[k * 20 + p]; akq = A[k * 20 + q]; }
        __syncthreads();
        if (k < 20) { A[k * 20 + p] = c * akp - s * akq; A[k * 20 + q] = s * akp + c * akq; }
        __syncthreads();
        float apk = 0, aqk = 0;
        if (k < 20) { apk = A[p * 20 + k]; aqk = A[q * 20 + k]; }
        __syncthreads();
        if (k < 20) {
          A[p * 20 + k] = c * apk - s * aqk;
          A[q * 20 + k] = s * apk + c * aqk;
          float vkp = V[k * 20 + p], vkq = V[k * 20 + q];
          V[k * 20 + p] = c * vkp - s * vkq;
          V[k * 20 + q] = s * vkp + c * vkq;
        }
        __syncthreads();
      }
    }
  }
}

__device__ void invsqrt_build(float* OUT, const float* A, const float* V, int lane) {
  __syncthreads();
  if (lane < 20) {
    int j = lane;
    for (int i = 0; i < 20; ++i) {
      float s = 0;
      for (int k = 0; k < 20; ++k) {
        float w = A[k * 20 + k];
        s += V[i * 20 + k] * V[j * 20 + k] * rsqrtf(w > 1e-30f ? w : 1e-30f);
      }
      OUT[i * 20 + j] = s;
    }
  }
  __syncthreads();
}

__device__ void mm20(float* C, const float* A, const float* B, int lane) {
  __syncthreads();
  if (lane < 20) {
    int j = lane;
    for (int i = 0; i < 20; ++i) {
      float s = 0;
      for (int k = 0; k < 20; ++k) s = fmaf(A[i * 20 + k], B[k * 20 + j], s);
      C[i * 20 + j] = s;
    }
  }
  __syncthreads();
}

__global__ __launch_bounds__(32) void k_solver(
    const float* __restrict__ P, float* __restrict__ out, float fm) {
  __shared__ float acc[1240];
  __shared__ float s12[400], A[400], V[400], U1[400], U2[400], T[400], TM[400];
  __shared__ float mug[20], muf[20], sc[2];
  const int lane = threadIdx.x;
  for (int e = lane; e < 1240; e += 32) {
    float s = 0;
    for (int b = 0; b < NSTATB; ++b) s += P[(size_t)b * 1240 + e];
    acc[e] = s;
  }
  __syncthreads();
  const float invden = 1.0f / (fm - 1.0f);
  if (lane < 20) { mug[lane] = acc[lane] / fm; muf[lane] = acc[20 + lane] / fm; }
  __syncthreads();
  if (lane < 20) {                       // s11 -> A (+reg), s12
    int j = lane;
    for (int i = 0; i < 20; ++i) {
      A[i * 20 + j]   = (acc[40 + i * 20 + j]  - fm * mug[i] * mug[j]) * invden
                        + ((i == j) ? REG1f : 0.0f);
      s12[i * 20 + j] = (acc[840 + i * 20 + j] - fm * mug[i] * muf[j]) * invden;
    }
  }
  if (lane == 0) {
    float vg = 0, vf = 0;
    for (int i = 0; i < 20; ++i) {
      vg += (acc[40 + i * 20 + i]  - fm * mug[i] * mug[i]) * invden;
      vf += (acc[440 + i * 20 + i] - fm * muf[i] * muf[i]) * invden;
    }
    sc[0] = vg / 20.0f;
    sc[1] = vf / 20.0f;
  }
  __syncthreads();
  jacobi20(A, V, lane);
  invsqrt_build(U1, A, V, lane);         // U1 = s11^{-1/2}
  if (lane < 20) {                       // s22 -> A (+reg)
    int j = lane;
    for (int i = 0; i < 20; ++i)
      A[i * 20 + j] = (acc[440 + i * 20 + j] - fm * muf[i] * muf[j]) * invden
                      + ((i == j) ? REG2f : 0.0f);
  }
  __syncthreads();
  jacobi20(A, V, lane);
  invsqrt_build(U2, A, V, lane);         // U2 = s22^{-1/2}
  mm20(TM, s12, U2, lane);               // TM = s12 @ U2
  mm20(T, U1, TM, lane);                 // T  = U1 @ TM
  if (lane < 20) {                       // A = T^T T
    int j = lane;
    for (int i = 0; i < 20; ++i) {
      float s = 0;
      for (int k = 0; k < 20; ++k) s = fmaf(T[k * 20 + i], T[k * 20 + j], s);
      A[i * 20 + j] = s;
    }
  }
  __syncthreads();
  jacobi20(A, V, lane);
  if (lane == 0) {                       // TOP_K == d: nuclear norm of T
    float s = 0;
    for (int k = 0; k < 20; ++k) {
      float w = A[k * 20 + k];
      s += sqrtf(w > 0.0f ? w : 0.0f);
    }
    out[0] = -s;
    out[1] = sc[0];
    out[2] = sc[1];
  }
}

// ------------------------------------------------------------------ host ----
extern "C" void kernel_launch(void* const* d_in, const int* in_sizes, int n_in,
                              void* d_out, int out_size, void* d_ws, size_t ws_size,
                              hipStream_t stream) {
  const float* node = (const float*)d_in[0];
  const float* emb  = (const float*)d_in[1];
  const float* W1   = (const float*)d_in[2];
  const float* b1   = (const float*)d_in[3];
  const float* W2   = (const float*)d_in[4];
  const float* b2   = (const float*)d_in[5];
  const float* Wih  = (const float*)d_in[6];
  // d_in[7] = W_hh: multiplied by h0 == 0 in the reference -> unused
  const float* bih  = (const float*)d_in[8];
  const float* bhh  = (const float*)d_in[9];
  const float* Wf   = (const float*)d_in[10];
  const float* bfv  = (const float*)d_in[11];
  const int*   ctx  = (const int*)d_in[12];

  const int m = in_sizes[0] / 1024;   // 65536
  const int L = in_sizes[12] / m;     // 32

  float* out = (float*)d_out;
  float* G = (float*)d_ws;                    // m x 20
  float* H = G + (size_t)m * 20;              // m x 20
  float* P = H + (size_t)m * 20;              // NSTATB x 1240 partials

  k_graph<<<dim3(m / 128), dim3(256), 0, stream>>>(node, W1, b1, W2, b2, G);
  k_lstm<<<dim3(m / 128), dim3(256), 0, stream>>>(emb, Wih, bih, bhh, ctx, L, H);
  k_ft<<<dim3((m * 64) / 256), dim3(256), 0, stream>>>(H, Wf, bfv, out + 3);
  k_stats<<<dim3(NSTATB), dim3(256), 0, stream>>>(G, H, P, m / NSTATB);
  k_solver<<<dim3(1), dim3(32), 0, stream>>>(P, out, (float)m);
}
